// Attention_66236985639131
// MI455X (gfx1250) — matrix operand order repaired and checked by reading, not compiled, not run on hardware
//
#include <hip/hip_runtime.h>
#include <hip/hip_bf16.h>

// ---------------------------------------------------------------------------
// MHA forward for B=2, N=2048, D=768, H=12, DH=64 on gfx1250 (CDNA5, wave32)
// f16 WMMA (v_wmma_f32_16x16x32_f16) with fp32 accumulation.
// Round 3: fixed base pointers + immediate offsets (no per-iter address VALU,
// no WMMA WAR-hazard nops), copy-free ping-pong double buffering in both the
// GEMM cores (fully unrolled K) and the flash-attention key loop.
// ---------------------------------------------------------------------------

#define BATCH 2
#define SEQ   2048
#define DIM   768
#define HEADS 12
#define DHEAD 64
#define MROWS (BATCH * SEQ)      // 4096
#define NQKV  (3 * DIM)          // 2304
#define PADH  4096               // pad (halves) after each ws buffer: dup
                                 // prefetches may read past the logical end

typedef _Float16 v16h __attribute__((ext_vector_type(16)));
typedef _Float16 v8h  __attribute__((ext_vector_type(8)));
typedef float    v8f  __attribute__((ext_vector_type(8)));

union F16x16 { v16h v; v8h h[2]; };

__device__ __forceinline__ v8f wmma_f16(v16h a, v16h b, v8f c) {
  // (neg_a, A, neg_b, B, c_mod, C, reuse_a, reuse_b)
  return __builtin_amdgcn_wmma_f32_16x16x32_f16(false, a, false, b, (short)0, c,
                                                false, false);
}

// A fragment from a per-lane base pointer: 16 contiguous halves.
__device__ __forceinline__ v16h frag_a(const _Float16* p) {
  F16x16 f;
  f.h[0] = *(const v8h*)(p);
  f.h[1] = *(const v8h*)(p + 8);
  return f.v;
}
// B fragment from a per-lane base pointer: 16 contiguous halves.
__device__ __forceinline__ v16h frag_b(const _Float16* p) {
  F16x16 f;
  f.h[0] = *(const v8h*)(p);
  f.h[1] = *(const v8h*)(p + 8);
  return f.v;
}

// Per-lane base offset for an A fragment (rows row0.., row-major, ld, k=0):
// lane m=l&15 holds row row0+m; K base (l>=16)*16.
__device__ __forceinline__ const _Float16* base_a(const _Float16* base, int row0,
                                                  int ld, int lane) {
  return base + (size_t)(row0 + (lane & 15)) * ld + ((lane >> 4) << 4);
}
// Per-lane base offset for a B fragment (cols n0.., column-major storage
// Bt[n][k], ld, k=0): lane n=l&15 holds col n0+n; K base (l>=16)*16.
__device__ __forceinline__ const _Float16* base_b(const _Float16* baseT, int n0,
                                                  int ld, int lane) {
  return baseT + (size_t)(n0 + (lane & 15)) * ld + ((lane >> 4) << 4);
}

// ---------------------------------------------------------------------------
// Kernel 1: fp32 -> f16 conversion; weights transposed for contiguous B-frags.
// ---------------------------------------------------------------------------
__global__ void cvt_kernel(const float* __restrict__ x,
                           const float* __restrict__ Wqkv,
                           const float* __restrict__ Wproj,
                           _Float16* __restrict__ xh,
                           _Float16* __restrict__ wqkvT,
                           _Float16* __restrict__ wprojT) {
  int i = blockIdx.x * blockDim.x + threadIdx.x;
  int stride = gridDim.x * blockDim.x;
  for (int idx = i; idx < MROWS * DIM; idx += stride)
    xh[idx] = (_Float16)x[idx];
  for (int idx = i; idx < NQKV * DIM; idx += stride) {
    int j = idx / DIM, k = idx - j * DIM;
    wqkvT[idx] = (_Float16)Wqkv[(size_t)k * NQKV + j];
  }
  for (int idx = i; idx < DIM * DIM; idx += stride) {
    int j = idx / DIM, k = idx - j * DIM;
    wprojT[idx] = (_Float16)Wproj[(size_t)k * DIM + j];
  }
}

// ---------------------------------------------------------------------------
// 32(M) x 64(N) x 768(K) wave-tile GEMM core. Fully unrolled K, fixed base
// pointers (all loads use immediate offsets), copy-free ping-pong buffers.
// acc[r][t]: r = row-tile (2 x 16 rows), t = col-tile (4 x 16 cols).
// ---------------------------------------------------------------------------
__device__ __forceinline__ void gemm_core_32x64(const _Float16* __restrict__ A,
                                                const _Float16* __restrict__ BT,
                                                int row0, int col0, int lane,
                                                v8f acc[2][4]) {
  const _Float16* pa[2];
  const _Float16* pb[4];
#pragma unroll
  for (int r = 0; r < 2; ++r) pa[r] = base_a(A, row0 + r * 16, DIM, lane);
#pragma unroll
  for (int t = 0; t < 4; ++t) pb[t] = base_b(BT, col0 + t * 16, DIM, lane);

  v16h aP[2], bP[4], aQ[2], bQ[4];
#pragma unroll
  for (int r = 0; r < 2; ++r) aP[r] = frag_a(pa[r]);
#pragma unroll
  for (int t = 0; t < 4; ++t) bP[t] = frag_b(pb[t]);

#pragma unroll
  for (int k0 = 0; k0 < DIM; k0 += 64) {
    // prefetch chunk k0+32 into Q while computing with P
#pragma unroll
    for (int r = 0; r < 2; ++r) aQ[r] = frag_a(pa[r] + k0 + 32);
#pragma unroll
    for (int t = 0; t < 4; ++t) bQ[t] = frag_b(pb[t] + k0 + 32);
#pragma unroll
    for (int r = 0; r < 2; ++r)
#pragma unroll
      for (int t = 0; t < 4; ++t)
        acc[r][t] = wmma_f16(aP[r], bP[t], acc[r][t]);

    // prefetch chunk k0+64 into P (last one reads buffer pad) with Q active
#pragma unroll
    for (int r = 0; r < 2; ++r) aP[r] = frag_a(pa[r] + k0 + 64);
#pragma unroll
    for (int t = 0; t < 4; ++t) bP[t] = frag_b(pb[t] + k0 + 64);
#pragma unroll
    for (int r = 0; r < 2; ++r)
#pragma unroll
      for (int t = 0; t < 4; ++t)
        acc[r][t] = wmma_f16(aQ[r], bQ[t], acc[r][t]);
  }
}

// ---------------------------------------------------------------------------
// Kernel 2: qkv = x @ W_qkv + b, scattered into attention layouts:
//   Q,K: [b,h,n,dh] (dh contiguous)   V: [b,h,dh,n] (n contiguous)
// Block: 256 thr = 8 waves; block tile 256(M) x 64(N); wave = 32x64.
// ---------------------------------------------------------------------------
__global__ void qkv_gemm(const _Float16* __restrict__ xh,
                         const _Float16* __restrict__ wT,
                         const float* __restrict__ bias,
                         _Float16* __restrict__ Qh,
                         _Float16* __restrict__ Kh,
                         _Float16* __restrict__ Vt) {
  int wave = threadIdx.x >> 5, lane = threadIdx.x & 31;
  int row0 = blockIdx.x * 256 + wave * 32;
  int col0 = blockIdx.y * 64;

  v8f acc[2][4] = {};
  gemm_core_32x64(xh, wT, row0, col0, lane, acc);

  int halfm = (lane >> 4) << 3;
  int nn = lane & 15;
#pragma unroll
  for (int r = 0; r < 2; ++r) {
#pragma unroll
    for (int t = 0; t < 4; ++t) {
#pragma unroll
      for (int v = 0; v < 8; ++v) {
        int row = row0 + r * 16 + halfm + v;     // [0,4096)
        int col = col0 + t * 16 + nn;            // [0,2304)
        float val = acc[r][t][v] + bias[col];
        int bb = row >> 11, n = row & (SEQ - 1);
        int which = col / DIM, cj = col - which * DIM;
        int h = cj >> 6, dh = cj & 63;
        _Float16 hv = (_Float16)val;
        size_t bh = (size_t)(bb * HEADS + h);
        if (which == 0)      Qh[(bh * SEQ + n) * DHEAD + dh] = hv;
        else if (which == 1) Kh[(bh * SEQ + n) * DHEAD + dh] = hv;
        else                 Vt[(bh * DHEAD + dh) * SEQ + n] = hv;
      }
    }
  }
}

// ---------------------------------------------------------------------------
// Kernel 3: flash attention. grid = B*H*(SEQ/64), block = 128 (4 waves),
// each wave owns 16 queries, streams keys 32 at a time. Pointer-increment
// addressing, ping-pong K fragments, V fragments prefetched before softmax.
// ---------------------------------------------------------------------------
__global__ void attn_kernel(const _Float16* __restrict__ Qh,
                            const _Float16* __restrict__ Kh,
                            const _Float16* __restrict__ Vt,
                            _Float16* __restrict__ Ah) {
  __shared__ _Float16 Pbuf[4][16][32];   // per-wave P tile (C->A transpose)

  int wave = threadIdx.x >> 5, lane = threadIdx.x & 31;
  int qt = blockIdx.x & 31;              // SEQ/64 = 32 query tiles
  int bh = blockIdx.x >> 5;              // b*H + h
  const _Float16* Qb = Qh + (size_t)bh * SEQ * DHEAD;
  const _Float16* Kb = Kh + (size_t)bh * SEQ * DHEAD;
  const _Float16* Vb = Vt + (size_t)bh * DHEAD * SEQ;
  int q0 = qt * 64 + wave * 16;

  // scale * log2(e): softmax in exp2 domain -> single v_exp_f32 per element
  const float qs = 0.125f * 1.44269504088896340736f;

  const _Float16* pq = base_a(Qb, q0, DHEAD, lane);
  v16h qf0 = frag_a(pq);
  v16h qf1 = frag_a(pq + 32);

  float mrow[8], lrow[8];
  v8f o[4] = {};
#pragma unroll
  for (int v = 0; v < 8; ++v) { mrow[v] = -3.0e38f; lrow[v] = 0.0f; }

  _Float16* myP = &Pbuf[wave][0][0];
  int halfm = (lane >> 4) << 3;
  int nn = lane & 15;

  // Per-lane K/V fragment pointers, advanced by constant strides.
  // K-frag order: [kt=0,dh0-31],[kt=1,dh0-31],[kt=0,dh32-63],[kt=1,dh32-63]
  const _Float16* kp[4];
  kp[0] = base_b(Kb, 0,  DHEAD, lane);
  kp[1] = base_b(Kb, 16, DHEAD, lane);
  kp[2] = kp[0] + 32;
  kp[3] = kp[1] + 32;
  const _Float16* vp[4];
#pragma unroll
  for (int t = 0; t < 4; ++t) vp[t] = base_b(Vt, 0, SEQ, lane) +
                                      (size_t)bh * DHEAD * SEQ + (size_t)(t * 16) * SEQ;

  v16h kfA[4], kfB[4];
#pragma unroll
  for (int i = 0; i < 4; ++i) { kfA[i] = frag_b(kp[i]); kp[i] += 32 * DHEAD; }

  // one 32-key step: uses kcur, prefetches knxt (last dup lands in pad)
  auto step32 = [&](v16h (&kcur)[4], v16h (&knxt)[4]) {
    v16h vf[4];
#pragma unroll
    for (int t = 0; t < 4; ++t) { vf[t] = frag_b(vp[t]); vp[t] += 32; }
#pragma unroll
    for (int i = 0; i < 4; ++i) { knxt[i] = frag_b(kp[i]); kp[i] += 32 * DHEAD; }

    // ---- S = Q @ K^T for 16q x 32k ----
    v8f s0 = {}, s1 = {};
    s0 = wmma_f16(qf0, kcur[0], s0);
    s1 = wmma_f16(qf0, kcur[1], s1);
    s0 = wmma_f16(qf1, kcur[2], s0);
    s1 = wmma_f16(qf1, kcur[3], s1);

    // ---- online softmax (rows live across 16-lane halves) ----
    float mnew[8], sc[8];
#pragma unroll
    for (int v = 0; v < 8; ++v) {
      float a = s0[v] * qs, b = s1[v] * qs;
      s0[v] = a; s1[v] = b;
      float mx = fmaxf(a, b);
      mx = fmaxf(mx, __shfl_xor(mx, 1, 32));
      mx = fmaxf(mx, __shfl_xor(mx, 2, 32));
      mx = fmaxf(mx, __shfl_xor(mx, 4, 32));
      mx = fmaxf(mx, __shfl_xor(mx, 8, 32));
      mnew[v] = fmaxf(mrow[v], mx);
      sc[v] = exp2f(mrow[v] - mnew[v]);
      mrow[v] = mnew[v];
    }
#pragma unroll
    for (int v = 0; v < 8; ++v) {
      float p0 = exp2f(s0[v] - mnew[v]);
      float p1 = exp2f(s1[v] - mnew[v]);
      s0[v] = p0; s1[v] = p1;
      float r = p0 + p1;
      r += __shfl_xor(r, 1, 32);
      r += __shfl_xor(r, 2, 32);
      r += __shfl_xor(r, 4, 32);
      r += __shfl_xor(r, 8, 32);
      lrow[v] = lrow[v] * sc[v] + r;
    }
#pragma unroll
    for (int t = 0; t < 4; ++t)
#pragma unroll
      for (int v = 0; v < 8; ++v) o[t][v] *= sc[v];

    // ---- transpose P (C layout -> A layout) through per-wave LDS ----
#pragma unroll
    for (int v = 0; v < 8; ++v) {
      myP[(halfm + v) * 32 + nn]      = (_Float16)s0[v];
      myP[(halfm + v) * 32 + 16 + nn] = (_Float16)s1[v];
    }
    asm volatile("s_wait_dscnt 0" ::: "memory");   // same-wave LDS in-order
    v16h pf = frag_a(myP + (lane & 15) * 32 + ((lane >> 4) << 4) - 0);
    // frag_a expects 16 contiguous halves at p: for the 16x32 P tile this is
    // exactly K[kb..kb+15] of row m.  (ld = 32)

    // ---- O += P @ V  (vf prefetched before the softmax) ----
#pragma unroll
    for (int t = 0; t < 4; ++t) o[t] = wmma_f16(pf, vf[t], o[t]);
  };

  for (int kb = 0; kb < SEQ; kb += 64) {
    step32(kfA, kfB);
    step32(kfB, kfA);
  }

  // ---- normalize and write concat-head output [b,n,d] in f16 ----
  int bb = bh / HEADS, h = bh - bb * HEADS;
#pragma unroll
  for (int v = 0; v < 8; ++v) {
    float inv = 1.0f / lrow[v];
    int row = q0 + halfm + v;
#pragma unroll
    for (int t = 0; t < 4; ++t) {
      Ah[((size_t)(bb * SEQ + row)) * DIM + h * DHEAD + t * 16 + nn] =
          (_Float16)(o[t][v] * inv);
    }
  }
}

// ---------------------------------------------------------------------------
// Kernel 4: out = attn @ W_proj + b_proj  (fp32 output)
// ---------------------------------------------------------------------------
__global__ void proj_gemm(const _Float16* __restrict__ Ah,
                          const _Float16* __restrict__ wT,
                          const float* __restrict__ bias,
                          float* __restrict__ out) {
  int wave = threadIdx.x >> 5, lane = threadIdx.x & 31;
  int row0 = blockIdx.x * 256 + wave * 32;
  int col0 = blockIdx.y * 64;

  v8f acc[2][4] = {};
  gemm_core_32x64(Ah, wT, row0, col0, lane, acc);

  int halfm = (lane >> 4) << 3;
  int nn = lane & 15;
#pragma unroll
  for (int r = 0; r < 2; ++r) {
#pragma unroll
    for (int t = 0; t < 4; ++t) {
#pragma unroll
      for (int v = 0; v < 8; ++v) {
        int row = row0 + r * 16 + halfm + v;
        int col = col0 + t * 16 + nn;
        out[(size_t)row * DIM + col] = acc[r][t][v] + bias[col];
      }
    }
  }
}

// ---------------------------------------------------------------------------
extern "C" void kernel_launch(void* const* d_in, const int* in_sizes, int n_in,
                              void* d_out, int out_size, void* d_ws,
                              size_t ws_size, hipStream_t stream) {
  (void)in_sizes; (void)n_in; (void)out_size; (void)ws_size;

  const float* x      = (const float*)d_in[0];
  const float* W_qkv  = (const float*)d_in[1];
  const float* b_qkv  = (const float*)d_in[2];
  const float* W_proj = (const float*)d_in[3];
  const float* b_proj = (const float*)d_in[4];
  float* out = (float*)d_out;

  _Float16* ws = (_Float16*)d_ws;
  const size_t n_x    = (size_t)MROWS * DIM;     // 3,145,728
  const size_t n_wq   = (size_t)NQKV * DIM;      // 1,769,472
  const size_t n_wp   = (size_t)DIM * DIM;       //   589,824
  const size_t n_head = (size_t)BATCH * HEADS * SEQ * DHEAD;  // 3,145,728

  _Float16* xh     = ws;
  _Float16* wqkvT  = xh + n_x + PADH;
  _Float16* wprojT = wqkvT + n_wq + PADH;
  _Float16* Qh     = wprojT + n_wp + PADH;
  _Float16* Kh     = Qh + n_head + PADH;
  _Float16* Vt     = Kh + n_head + PADH;
  _Float16* Ah     = Vt + n_head + PADH;         // total ~36.3 MB incl pads

  cvt_kernel<<<2048, 256, 0, stream>>>(x, W_qkv, W_proj, xh, wqkvT, wprojT);

  dim3 g1(MROWS / 256, NQKV / 64);               // 16 x 36
  qkv_gemm<<<g1, 256, 0, stream>>>(xh, wqkvT, b_qkv, Qh, Kh, Vt);

  attn_kernel<<<BATCH * HEADS * (SEQ / 64), 128, 0, stream>>>(Qh, Kh, Vt, Ah);

  dim3 g2(MROWS / 256, DIM / 64);                // 16 x 12
  proj_gemm<<<g2, 256, 0, stream>>>(Ah, wprojT, b_proj, out);
}
